// LocalityAttention_37830071943540
// MI455X (gfx1250) — matrix operand order repaired and checked by reading, not compiled, not run on hardware
//
#include <hip/hip_runtime.h>
#include <cstdint>
#include <cstddef>

// ---------------------------------------------------------------------------
// LocalityAttention for MI455X (gfx1250): bf16 WMMA flash attention.
//   B=32, S=1024, D=768.  Q=qWq+b, K=kWk+b, V=vWv+b (bf16 in ws),
//   O = softmax(QK^T/T, diag=-inf) V   (fp32 out)
// ---------------------------------------------------------------------------

#define B_ 32
#define S_ 1024
#define D_ 768

typedef __attribute__((ext_vector_type(16))) __bf16 bf16x16;
typedef __attribute__((ext_vector_type(8)))  __bf16 bf16x8;
typedef __attribute__((ext_vector_type(4)))  __bf16 bf16x4;
typedef __attribute__((ext_vector_type(8)))  float  f32x8;

__device__ __forceinline__ __bf16 f2bf(float f) {
  union { float f; unsigned u; } a; a.f = f;
  unsigned r = a.u + 0x7FFFu + ((a.u >> 16) & 1u);   // round-to-nearest-even
  union { unsigned short s; __bf16 b; } o; o.s = (unsigned short)(r >> 16);
  return o.b;
}

__device__ __forceinline__ f32x8 wmma_bf16(bf16x16 a, bf16x16 b, f32x8 c) {
#if defined(__HIP_DEVICE_COMPILE__)
  // 8 args: (neg_a, A, neg_b, B, c_mod, C, reuse_a, reuse_b)
  return __builtin_amdgcn_wmma_f32_16x16x32_bf16(false, a, false, b,
                                                 (short)0, c, false, false);
#else
  return c;
#endif
}

// 16-byte global -> LDS copy via the CDNA5 async-to-LDS path (inline asm:
// ROCm 7.2's clang builtin has a non-upstream signature, so bypass it).
// global_load_async_to_lds_b128 vDST(lds byte offset), vADDR(64b global), off
__device__ __forceinline__ void cp16(void* lds, const void* gp) {
#if defined(__gfx1250__)
  unsigned loff = (unsigned)(unsigned long long)
      (__attribute__((address_space(3))) char*)lds;
  asm volatile("global_load_async_to_lds_b128 %0, %1, off"
               :: "v"(loff), "v"((unsigned long long)gp)
               : "memory");
#else
  *(uint4*)lds = *(const uint4*)gp;
#endif
}

__device__ __forceinline__ void cp_wait() {
#if defined(__gfx1250__)
  asm volatile("s_wait_asynccnt 0" ::: "memory");
#endif
}

// ---------------------------------------------------------------------------
// Projection: out[b, s, n] (or transposed out[b, n, s]) =
//             sum_k X[b, s, k] * W[k, n] + bias[n]    (stored as bf16)
// Block: 256 threads = 8 waves; tile 128 rows x 64 cols; K-chunks of 32.
// Wave w owns a 32x32 quadrant (wr = w&3, wc = w>>2) = 2x2 WMMA tiles,
// so 4 WMMAs per K-step with 2 A-frags x 2 B-frags reuse.
// ---------------------------------------------------------------------------
#define PT_M 128
#define PT_N 64
#define PT_K 32

__global__ __launch_bounds__(256, 1)
void proj_bf16_kernel(const float* __restrict__ X, const float* __restrict__ W,
                      const float* __restrict__ bias, __bf16* __restrict__ out,
                      int transposed)
{
  __shared__ __align__(32) __bf16 lsA[PT_M * PT_K];   // [m][k]      8 KB
  __shared__ __align__(32) __bf16 lsW[PT_N * PT_K];   // [n][k] (T)  4 KB

  const int tid  = threadIdx.x;
  const int lane = tid & 31;
  const int wv   = tid >> 5;
  const int rb   = blockIdx.x * PT_M;
  const int cb   = blockIdx.y * PT_N;
  const int b    = blockIdx.z;
  const int wr   = wv & 3;     // 4 row groups of 32
  const int wc   = wv >> 2;    // 2 col groups of 32
  const int kh   = lane >> 4;
  const int l15  = lane & 15;

  const f32x8 vzero = {0.f, 0.f, 0.f, 0.f, 0.f, 0.f, 0.f, 0.f};
  f32x8 pacc[2][2];
#pragma unroll
  for (int sr = 0; sr < 2; ++sr)
#pragma unroll
    for (int sc = 0; sc < 2; ++sc) pacc[sr][sc] = vzero;

  for (int kc = 0; kc < D_; kc += PT_K) {
    // A tile 128x32: float4 per thread per trip (4 trips)
    for (int i = tid; i < (PT_M * PT_K) / 4; i += 256) {
      int r = i >> 3, c4 = (i & 7) * 4;
      float4 x4 = *(const float4*)&X[((size_t)b * S_ + rb + r) * D_ + kc + c4];
      bf16x4 t;
      t[0] = f2bf(x4.x); t[1] = f2bf(x4.y); t[2] = f2bf(x4.z); t[3] = f2bf(x4.w);
      *(bf16x4*)&lsA[r * PT_K + c4] = t;
    }
    // W tile 32x64 -> transposed [n][k]: float4 per thread per trip (2 trips)
    for (int i = tid; i < (PT_K * PT_N) / 4; i += 256) {
      int k = i >> 4, n4 = (i & 15) * 4;
      float4 w4 = *(const float4*)&W[(size_t)(kc + k) * D_ + cb + n4];
      lsW[(n4 + 0) * PT_K + k] = f2bf(w4.x);
      lsW[(n4 + 1) * PT_K + k] = f2bf(w4.y);
      lsW[(n4 + 2) * PT_K + k] = f2bf(w4.z);
      lsW[(n4 + 3) * PT_K + k] = f2bf(w4.w);
    }
    __syncthreads();

    bf16x16 afr[2], bfr[2];
#pragma unroll
    for (int sr = 0; sr < 2; ++sr) {
      const int m = wr * 32 + sr * 16 + l15;
      union { bf16x16 v; bf16x8 h[2]; } ua;
      ua.h[0] = *(const bf16x8*)&lsA[m * PT_K + kh * 16];
      ua.h[1] = *(const bf16x8*)&lsA[m * PT_K + kh * 16 + 8];
      afr[sr] = ua.v;
    }
#pragma unroll
    for (int sc = 0; sc < 2; ++sc) {
      const int n = wc * 32 + sc * 16 + l15;
      bfr[sc] = *(const bf16x16*)&lsW[n * PT_K + kh * 16];
    }
#pragma unroll
    for (int sr = 0; sr < 2; ++sr)
#pragma unroll
      for (int sc = 0; sc < 2; ++sc)
        pacc[sr][sc] = wmma_bf16(afr[sr], bfr[sc], pacc[sr][sc]);
    __syncthreads();
  }

#pragma unroll
  for (int sr = 0; sr < 2; ++sr)
#pragma unroll
    for (int sc = 0; sc < 2; ++sc) {
      const int ng    = cb + wc * 32 + sc * 16 + l15;
      const float bb  = bias[ng];
      const int mrow0 = rb + wr * 32 + sr * 16 + kh * 8;
      if (!transposed) {                   // row-major [S][D] (Q, K)
#pragma unroll
        for (int r = 0; r < 8; ++r)
          out[((size_t)b * S_ + mrow0 + r) * D_ + ng] =
              f2bf(pacc[sr][sc][r] + bb);
      } else {                             // transposed [D][S] (V)
        bf16x8 o;
#pragma unroll
        for (int r = 0; r < 8; ++r) o[r] = f2bf(pacc[sr][sc][r] + bb);
        *(bf16x8*)&out[((size_t)b * D_ + ng) * S_ + mrow0] = o;
      }
    }
}

// ---------------------------------------------------------------------------
// Flash attention: block = 256 thr (8 waves), 32 query rows per block.
// Key chunks of 32.  Waves 0-3 compute the 32x32 score block; every lane
// (lane==row) runs online softmax; each wave owns a 96-wide d-slab of O.
// ---------------------------------------------------------------------------
#define QT 32
#define KT 32
#define ATTN_SMEM (QT*D_*2 + KT*D_*2 + QT*KT*4 + QT*KT*2 + 32*4 + 32*4)

__global__ __launch_bounds__(256, 1)
void flash_attn_kernel(const __bf16* __restrict__ Qb,
                       const __bf16* __restrict__ Kb,
                       const __bf16* __restrict__ Vt,   // [B][D][S]
                       const float*  __restrict__ temp,
                       float* __restrict__ out)
{
  extern __shared__ __align__(32) char smem[];
  __bf16* sQ = (__bf16*)smem;            // QT x D_
  __bf16* sK = sQ + QT * D_;             // KT x D_
  float*  sS = (float*)(sK + KT * D_);   // QT x KT scores
  __bf16* sP = (__bf16*)(sS + QT * KT);  // QT x KT probs (bf16)
  float*  sC = (float*)(sP + QT * KT);   // 32 rescale factors
  float*  sL = sC + 32;                  // 32 final row sums

  const int tid  = threadIdx.x;
  const int lane = tid & 31;
  const int wv   = tid >> 5;
  const int q0   = blockIdx.x * QT;
  const int b    = blockIdx.y;
  const float invT = 1.0f / temp[0];

  // ---- stage Q tile (resident for the whole kernel) ----
  {
    const char* g = (const char*)(Qb + ((size_t)b * S_ + q0) * D_);
    char* l = (char*)sQ;
    for (int i = tid; i < (QT * D_) / 8; i += 256) cp16(l + i * 16, g + i * 16);
    cp_wait();
  }
  __syncthreads();

  const f32x8 vzero = {0.f, 0.f, 0.f, 0.f, 0.f, 0.f, 0.f, 0.f};
  f32x8 acc[2][6];
#pragma unroll
  for (int qs = 0; qs < 2; ++qs)
#pragma unroll
    for (int t6 = 0; t6 < 6; ++t6) acc[qs][t6] = vzero;

  float m_run = -3.0e38f, l_run = 0.f;
  const int sgq = q0 + lane;      // this lane's stat row (global query idx)
  const int kh  = lane >> 4;

  for (int kc = 0; kc < S_ / KT; ++kc) {
    // ---- stage K chunk ----
    {
      const char* g = (const char*)(Kb + ((size_t)b * S_ + kc * KT) * D_);
      char* l = (char*)sK;
      for (int i = tid; i < (KT * D_) / 8; i += 256) cp16(l + i * 16, g + i * 16);
      cp_wait();
    }
    __syncthreads();

    // ---- scores: waves 0-3, one 16x16 tile each ----
    if (wv < 4) {
      const int qsub = wv >> 1, ksub = wv & 1;
      const int sm = qsub * 16 + (lane & 15);
      const int st = ksub * 16 + (lane & 15);
      f32x8 sAcc = vzero;
      for (int dk = 0; dk < D_; dk += 32) {
        union { bf16x16 v; bf16x8 h[2]; } ua;
        ua.h[0] = *(const bf16x8*)&sQ[sm * D_ + dk + kh * 16];
        ua.h[1] = *(const bf16x8*)&sQ[sm * D_ + dk + kh * 16 + 8];
        bf16x16 bv = *(const bf16x16*)&sK[st * D_ + dk + kh * 16];
        sAcc = wmma_bf16(ua.v, bv, sAcc);
      }
#pragma unroll
      for (int r = 0; r < 8; ++r)
        sS[(qsub * 16 + kh * 8 + r) * KT + ksub * 16 + (lane & 15)] = sAcc[r];
    }
    __syncthreads();

    // ---- online softmax: lane == query row (redundant per wave) ----
    {
      float sc[32];
#pragma unroll
      for (int j4 = 0; j4 < 8; ++j4) {
        float4 v4 = *(const float4*)&sS[lane * KT + j4 * 4];
        sc[4 * j4 + 0] = v4.x; sc[4 * j4 + 1] = v4.y;
        sc[4 * j4 + 2] = v4.z; sc[4 * j4 + 3] = v4.w;
      }
      float mx = m_run;
#pragma unroll
      for (int j = 0; j < 32; ++j) {
        float s = sc[j] * invT;
        if (kc * KT + j == sgq) s = -3.0e38f;   // diagonal mask
        sc[j] = s;
        mx = fmaxf(mx, s);
      }
      const float corr = __expf(m_run - mx);
      float sum = 0.f;
      __bf16 pb[32];
#pragma unroll
      for (int j = 0; j < 32; ++j) {
        float p = __expf(sc[j] - mx);
        sum += p;
        pb[j] = f2bf(p);
      }
      l_run = l_run * corr + sum;
      m_run = mx;
      if (wv == 0) {
#pragma unroll
        for (int g8 = 0; g8 < 4; ++g8) {
          bf16x8 t8;
#pragma unroll
          for (int e = 0; e < 8; ++e) t8[e] = pb[g8 * 8 + e];
          *(bf16x8*)&sP[lane * KT + g8 * 8] = t8;
        }
        sC[lane] = corr;
      }
    }
    __syncthreads();

    // ---- rescale O and accumulate P * V (all 8 waves, 96-col slab each) ----
    {
      float cg[2][8];
#pragma unroll
      for (int qs = 0; qs < 2; ++qs)
#pragma unroll
        for (int r = 0; r < 8; ++r)
          cg[qs][r] = sC[qs * 16 + kh * 8 + r];
#pragma unroll
      for (int qs = 0; qs < 2; ++qs)
#pragma unroll
        for (int t6 = 0; t6 < 6; ++t6)
#pragma unroll
          for (int r = 0; r < 8; ++r)
            acc[qs][t6][r] *= cg[qs][r];

      bf16x16 pa[2];
#pragma unroll
      for (int qs = 0; qs < 2; ++qs) {
        const int pm = qs * 16 + (lane & 15);
        union { bf16x16 v; bf16x8 h[2]; } up;
        up.h[0] = *(const bf16x8*)&sP[pm * KT + kh * 16];
        up.h[1] = *(const bf16x8*)&sP[pm * KT + kh * 16 + 8];
        pa[qs] = up.v;
      }
#pragma unroll
      for (int t6 = 0; t6 < 6; ++t6) {
        const int n = wv * 96 + t6 * 16 + (lane & 15);
        const __bf16* vp = Vt + ((size_t)b * D_ + n) * S_ + kc * KT + kh * 16;
        bf16x16 bv = *(const bf16x16*)vp;
#pragma unroll
        for (int qs = 0; qs < 2; ++qs)
          acc[qs][t6] = wmma_bf16(pa[qs], bv, acc[qs][t6]);
      }
    }
    __syncthreads();
  }

  if (wv == 0) sL[lane] = l_run;
  __syncthreads();

  float linv[2][8];
#pragma unroll
  for (int qs = 0; qs < 2; ++qs)
#pragma unroll
    for (int r = 0; r < 8; ++r)
      linv[qs][r] = 1.0f / sL[qs * 16 + kh * 8 + r];

#pragma unroll
  for (int qs = 0; qs < 2; ++qs)
#pragma unroll
    for (int t6 = 0; t6 < 6; ++t6) {
      const int n = wv * 96 + t6 * 16 + (lane & 15);
#pragma unroll
      for (int r = 0; r < 8; ++r)
        out[((size_t)b * S_ + q0 + qs * 16 + kh * 8 + r) * D_ + n] =
            acc[qs][t6][r] * linv[qs][r];
    }
}

// ---------------------------------------------------------------------------
extern "C" void kernel_launch(void* const* d_in, const int* in_sizes, int n_in,
                              void* d_out, int out_size, void* d_ws, size_t ws_size,
                              hipStream_t stream) {
  (void)in_sizes; (void)n_in; (void)out_size; (void)ws_size;
  const float* q    = (const float*)d_in[0];
  const float* k    = (const float*)d_in[1];
  const float* v    = (const float*)d_in[2];
  const float* Wq   = (const float*)d_in[3];
  const float* bq   = (const float*)d_in[4];
  const float* Wk   = (const float*)d_in[5];
  const float* bk   = (const float*)d_in[6];
  const float* Wv   = (const float*)d_in[7];
  const float* bv   = (const float*)d_in[8];
  const float* temp = (const float*)d_in[9];
  float* out = (float*)d_out;

  const size_t tensorElems = (size_t)B_ * S_ * D_;
  __bf16* Qb = (__bf16*)d_ws;
  __bf16* Kb = Qb + tensorElems;
  __bf16* Vt = Kb + tensorElems;   // transposed [B][D][S]

  dim3 pg(S_ / PT_M, D_ / PT_N, B_);
  proj_bf16_kernel<<<pg, 256, 0, stream>>>(q, Wq, bq, Qb, 0);
  proj_bf16_kernel<<<pg, 256, 0, stream>>>(k, Wk, bk, Kb, 0);
  proj_bf16_kernel<<<pg, 256, 0, stream>>>(v, Wv, bv, Vt, 1);

  dim3 ag(S_ / QT, B_);
  flash_attn_kernel<<<ag, 256, (size_t)ATTN_SMEM, stream>>>(Qb, Kb, Vt, temp, out);
}
